// GradientLoss_28063316312224
// MI455X (gfx1250) — compile-verified, matrix operand order checked by reading, not run on hardware
//
#include <hip/hip_runtime.h>
#include <hip/hip_bf16.h>

typedef __attribute__((ext_vector_type(16))) _Float16 v16h;
typedef __attribute__((ext_vector_type(8)))  float    v8f;

#define HW  (1024 * 1024)
#define WID 1024
#define NCLS 4

// Exact 32-lane sum via V_WMMA_F32_16X16X32_F16 with a ones B-matrix.
// A: lane L contributes at (M = L%16, K = L<16 ? 0 : 8)  -> D[m][*] = v[m] + v[m+16].
// All columns of D are identical; d[0..7] on lanes 0-15 cover M=0..7, lanes 16-31 M=8..15.
// Inputs are small non-negative integers (<= 32), exact in f16; accumulation is f32.
__device__ __forceinline__ float waveReduceWMMA(float v) {
    v16h a = {};
    a[0] = (_Float16)v;
    v16h b;
#pragma unroll
    for (int i = 0; i < 16; ++i) b[i] = (_Float16)1.0f;
    v8f c = {};
    v8f d = __builtin_amdgcn_wmma_f32_16x16x32_f16(
        /*neg_a=*/false, a, /*neg_b=*/false, b,
        /*c_mod=*/(short)0, c, /*reuse_a=*/false, /*reuse_b=*/false);
    float s = d[0] + d[1] + d[2] + d[3] + d[4] + d[5] + d[6] + d[7];
    s += __shfl_xor(s, 16, 32);   // add other half's row-sums -> full 32-lane total
    return s;
}

__global__ void gl_init(float* acc) {
    if (threadIdx.x < 3 * NCLS) acc[threadIdx.x] = 0.0f;
}

__global__ __launch_bounds__(256) void gl_main(const float* __restrict__ edge,
                                               const int* __restrict__ gt,
                                               float* __restrict__ acc, int B) {
    float accE[NCLS] = {0, 0, 0, 0};
    float accG[NCLS] = {0, 0, 0, 0};
    float accI[NCLS] = {0, 0, 0, 0};

    int tid    = blockIdx.x * blockDim.x + threadIdx.x;
    int stride = gridDim.x * blockDim.x;

    for (int p = tid; p < HW; p += stride) {
        int h  = p >> 10;
        int w  = p & 1023;
        int sy = (h * 1022) >> 10;           // nearest-resize source row (conv-output coords)
        int sx = (w * 1022) >> 10;

        int eCnt[NCLS] = {0, 0, 0, 0};       // Sum_b e[b,c,p]
        int gCnt[NCLS] = {0, 0, 0, 0};       // Sum_b g[b,c,p]

        for (int b = 0; b < B; ++b) {
            // ---- e: 2-way softmax argmax per class pair ----
            const float* eb = edge + (size_t)b * (2 * NCLS) * HW + p;
#pragma unroll
            for (int cI = 0; cI < NCLS; ++cI) {
                float v0 = eb[(size_t)(2 * cI)     * HW];
                float v1 = eb[(size_t)(2 * cI + 1) * HW];
                eCnt[cI] += (v1 > v0) ? 1 : 0;
            }
            // ---- g: binarized Sobel on one-hot(gt); center tap has weight 0 in both ----
            const int* gb = gt + (size_t)b * HW + sy * WID + sx;
            int a0 = gb[0],        a1 = gb[1],           a2 = gb[2];
            int a3 = gb[WID],                            a5 = gb[WID + 2];
            int a6 = gb[2 * WID],  a7 = gb[2 * WID + 1], a8 = gb[2 * WID + 2];
#pragma unroll
            for (int cI = 0; cI < NCLS; ++cI) {
                int i0 = (a0 == cI), i1 = (a1 == cI), i2 = (a2 == cI);
                int i3 = (a3 == cI),                  i5 = (a5 == cI);
                int i6 = (a6 == cI), i7 = (a7 == cI), i8 = (a8 == cI);
                int gx = (i2 - i0) + 2 * (i5 - i3) + (i8 - i6);
                int gy = (i6 - i0) + 2 * (i7 - i1) + (i8 - i2);
                gCnt[cI] += ((gx | gy) != 0) ? 1 : 0;
            }
        }
#pragma unroll
        for (int cI = 0; cI < NCLS; ++cI) {
            float fe = (float)eCnt[cI];
            float fg = (float)gCnt[cI];
            accE[cI] += fe;
            accG[cI] += fg;
            accI[cI] += fe * fg;             // per-pixel batch-sum product
        }
    }

    // ---- wave reduction via WMMA, then block reduction via LDS ----
    __shared__ float red[8 * 3 * NCLS];
    int lane = threadIdx.x & 31;
    int wv   = threadIdx.x >> 5;
#pragma unroll
    for (int cI = 0; cI < NCLS; ++cI) {
        float se = waveReduceWMMA(accE[cI]);
        float sg = waveReduceWMMA(accG[cI]);
        float si = waveReduceWMMA(accI[cI]);
        if (lane == 0) {
            red[wv * 12 + cI]     = se;
            red[wv * 12 + 4 + cI] = sg;
            red[wv * 12 + 8 + cI] = si;
        }
    }
    __syncthreads();
    if (threadIdx.x < 12) {
        float s = 0.0f;
#pragma unroll
        for (int w8 = 0; w8 < 8; ++w8) s += red[w8 * 12 + threadIdx.x];
        atomicAdd(&acc[threadIdx.x], s);
    }
}

__global__ void gl_finalize(const float* __restrict__ acc, float* __restrict__ out, float Bf) {
    if (threadIdx.x == 0) {
        float loss = 0.0f;
#pragma unroll
        for (int cI = 0; cI < NCLS; ++cI) {
            float Se = acc[cI], Sg = acc[4 + cI], Si = acc[8 + cI];
            float inter = 2.0f * Si;
            float denom = Bf * Se + Bf * Sg + 1e-10f;
            loss += 1.0f - inter / denom;
        }
        out[0] = loss / (float)NCLS;
    }
}

extern "C" void kernel_launch(void* const* d_in, const int* in_sizes, int n_in,
                              void* d_out, int out_size, void* d_ws, size_t ws_size,
                              hipStream_t stream) {
    const float* edge = (const float*)d_in[0];
    const int*   gt   = (const int*)d_in[1];
    float* acc = (float*)d_ws;                    // 12 floats of scratch
    int B = in_sizes[0] / (2 * NCLS * HW);        // = 4 for the reference shapes

    gl_init<<<1, 32, 0, stream>>>(acc);
    gl_main<<<2048, 256, 0, stream>>>(edge, gt, acc, B);
    gl_finalize<<<1, 32, 0, stream>>>(acc, (float*)d_out, (float)B);
}
